// LowRank2d_76562087019199
// MI455X (gfx1250) — matrix operand order repaired and checked by reading, not compiled, not run on hardware
//
#include <hip/hip_runtime.h>
#include <cstddef>

// ---------------------------------------------------------------------------
// Problem constants (from reference)
// ---------------------------------------------------------------------------
#define BATCH   10
#define NPTS    7225
#define WIDTH   32
#define NCHUNK  113          // ceil(7225/64)
#define NSEG    32           // segments for the psi reduction GEMM

typedef __attribute__((ext_vector_type(16))) __bf16         v16bf;
typedef __attribute__((ext_vector_type(8)))  float          v8f;
typedef __attribute__((ext_vector_type(8)))  unsigned short ush8;

// ---------------------------------------------------------------------------
// Helpers
// ---------------------------------------------------------------------------
__device__ __forceinline__ unsigned short f2bf(float x) {
    // native f32->bf16 (RNE) — gfx1250 has hardware bf16 converts
    union { __bf16 b; unsigned short u; } c;
    c.b = (__bf16)x;
    return c.u;
}

union FragU { ush8 h[2]; v16bf v; };

// A fragment: 16x32 bf16, source stored row-major [M][ldk] (raw bf16 bits).
// base points at (row, k0). Lane layout: lane&15 = M row; lane half selects
// K sub-blocks {half*8..half*8+7} and {16+half*8..16+half*8+7}.
__device__ __forceinline__ v16bf load_frag_A(const unsigned short* base, int half) {
    FragU u;
    u.h[0] = *(const ush8*)(base + half * 8);
    u.h[1] = *(const ush8*)(base + 16 + half * 8);
    return u.v;
}

// B fragment: 32x16 bf16 (KxN), source stored N-major (transposed) [N][ldk].
// base points at (col, k0). Lane layout: lane&15 = N col; lane half selects
// K sub-blocks {half*8..half*8+7} and {16+half*8..16+half*8+7}.
__device__ __forceinline__ v16bf load_frag_B(const unsigned short* base, int half) {
    FragU u;
    u.h[0] = *(const ush8*)(base + half * 8);
    u.h[1] = *(const ush8*)(base + 16 + half * 8);
    return u.v;
}

__device__ __forceinline__ v8f wmma_bf16(v16bf a, v16bf b, v8f c) {
    return __builtin_amdgcn_wmma_f32_16x16x32_bf16(false, a, false, b,
                                                   (short)0, c, false, false);
}

// ---------------------------------------------------------------------------
// Fused 3-layer MLP (3 -> 64 -> 128 -> 256, ReLU after each) for a 64-point
// tile, 256 threads (8 wave32s). Hidden activations live in LDS as bf16.
// H3MODE 0: hs3 stored [m][256] row-major.  H3MODE 1: hs3 stored [k][64].
// ---------------------------------------------------------------------------
template<int H3MODE>
__device__ __forceinline__ void mlp3_tile(
    const float* __restrict__ a_g,      // &a[(b*NPTS + n0)*3]
    int P,                              // valid points in tile (<=64)
    const float* __restrict__ w1, const float* __restrict__ b1,
    const unsigned short* __restrict__ w2t, const float* __restrict__ b2,
    const unsigned short* __restrict__ w3t, const float* __restrict__ b3,
    float* a_s, unsigned short* hs1, unsigned short* hs2, unsigned short* hs3)
{
    const int tid  = threadIdx.x;
    const int wave = tid >> 5;
    const int lane = tid & 31;
    const int half = lane >> 4;
    const int lm   = lane & 15;

    // stage inputs (zero-pad past P)
    if (tid < 192) {
        int p = tid / 3;
        a_s[tid] = (p < P) ? a_g[tid] : 0.0f;
    }
    __syncthreads();

    // Layer 1: 3 -> 64 (scalar VALU, trivial)
    for (int e = tid; e < 64 * 64; e += 256) {
        int p = e >> 6, j = e & 63;
        float s = b1[j];
        s += a_s[p * 3 + 0] * w1[0 * 64 + j];
        s += a_s[p * 3 + 1] * w1[1 * 64 + j];
        s += a_s[p * 3 + 2] * w1[2 * 64 + j];
        hs1[e] = f2bf(fmaxf(s, 0.0f));
    }
    __syncthreads();

    // Layer 2: 64x64 @ 64x128 -> 64x128 ; 4x8 tiles, 4 per wave, K-steps=2
    for (int q = 0; q < 4; ++q) {
        int f = wave * 4 + q;
        int mi = f >> 3, ni = f & 7;
        v8f acc = {};
        for (int ks = 0; ks < 2; ++ks) {
            v16bf A = load_frag_A(hs1 + (mi * 16 + lm) * 64 + ks * 32, half);
            v16bf B = load_frag_B(w2t + (ni * 16 + lm) * 64 + ks * 32, half);
            acc = wmma_bf16(A, B, acc);
        }
        int o = ni * 16 + lm;
        float bo = b2[o];
        for (int j = 0; j < 8; ++j) {
            int m = mi * 16 + half * 8 + j;
            hs2[m * 128 + o] = f2bf(fmaxf(acc[j] + bo, 0.0f));
        }
    }
    __syncthreads();

    // Layer 3: 64x128 @ 128x256 -> 64x256 ; 4x16 tiles, 8 per wave, K-steps=4
    for (int q = 0; q < 8; ++q) {
        int f = wave * 8 + q;
        int mi = f >> 4, ni = f & 15;
        v8f acc = {};
        for (int ks = 0; ks < 4; ++ks) {
            v16bf A = load_frag_A(hs2 + (mi * 16 + lm) * 128 + ks * 32, half);
            v16bf B = load_frag_B(w3t + (ni * 16 + lm) * 128 + ks * 32, half);
            acc = wmma_bf16(A, B, acc);
        }
        int o = ni * 16 + lm;
        float bo = b3[o];
        for (int j = 0; j < 8; ++j) {
            int m = mi * 16 + half * 8 + j;
            unsigned short hv = f2bf(fmaxf(acc[j] + bo, 0.0f));
            if (H3MODE == 0) hs3[m * 256 + o] = hv;   // row-major [m][k]
            else             hs3[o * 64 + m]  = hv;   // transposed [k][m]
        }
    }
    __syncthreads();
}

// ---------------------------------------------------------------------------
// K0: prep — transpose+bf16 weights, Wsum/bsum (phi last layer folded over i),
//     Vs[b][i] = sum_n v[b][n][i]
// ---------------------------------------------------------------------------
__global__ __launch_bounds__(256) void k0_prep(
    const float* __restrict__ phi_w2, const float* __restrict__ psi_w2,
    const float* __restrict__ phi_w3, const float* __restrict__ psi_w3,
    const float* __restrict__ phi_w4, const float* __restrict__ phi_b4,
    const float* __restrict__ v,
    unsigned short* __restrict__ w2t_phi, unsigned short* __restrict__ w2t_psi,
    unsigned short* __restrict__ w3t_phi, unsigned short* __restrict__ w3t_psi,
    float* __restrict__ wsum, float* __restrict__ bsum, float* __restrict__ Vs)
{
    int e = blockIdx.x * 256 + threadIdx.x;
    if (e < 8192) {                                   // phi W2 (64x128) -> [n][k]
        int k = e >> 7, n = e & 127;
        w2t_phi[n * 64 + k] = f2bf(phi_w2[e]);
    } else if (e < 16384) { e -= 8192;                // psi W2
        int k = e >> 7, n = e & 127;
        w2t_psi[n * 64 + k] = f2bf(psi_w2[e]);
    } else if (e < 49152) { e -= 16384;               // phi W3 (128x256) -> [n][k]
        int k = e >> 8, n = e & 255;
        w3t_phi[n * 128 + k] = f2bf(phi_w3[e]);
    } else if (e < 81920) { e -= 49152;               // psi W3
        int k = e >> 8, n = e & 255;
        w3t_psi[n * 128 + k] = f2bf(psi_w3[e]);
    } else if (e < 98304) { e -= 81920;               // Wsum[k][o*2+r] = sum_i W4phi
        int k = e >> 6, orr = e & 63;
        int o = orr >> 1, r = orr & 1;
        float s = 0.0f;
        for (int i = 0; i < 32; ++i) s += phi_w4[k * 2048 + o * 64 + i * 2 + r];
        wsum[e] = s;
    } else if (e < 98368) { e -= 98304;               // bsum[o*2+r]
        int o = e >> 1, r = e & 1;
        float s = 0.0f;
        for (int i = 0; i < 32; ++i) s += phi_b4[o * 64 + i * 2 + r];
        bsum[e] = s;
    } else if (e < 98688) { e -= 98368;               // Vs[b][i]
        int b = e >> 5, i = e & 31;
        float s = 0.0f;
        for (int n = 0; n < NPTS; ++n) s += v[((size_t)b * NPTS + n) * 32 + i];
        Vs[e] = s;
    }
}

// ---------------------------------------------------------------------------
// K2: fused psi MLP3 + G_part[seg][b][k][i] = sum_{n in seg} g3[n][k]*v[n][i]
// grid (NSEG, BATCH), 256 threads. Accumulators kept in VGPRs across chunks.
// ---------------------------------------------------------------------------
__global__ __launch_bounds__(256) void k2_psi(
    const float* __restrict__ v, const float* __restrict__ a,
    const float* __restrict__ w1, const float* __restrict__ b1,
    const unsigned short* __restrict__ w2t, const float* __restrict__ b2,
    const unsigned short* __restrict__ w3t, const float* __restrict__ b3,
    float* __restrict__ Gpart)
{
    __shared__ __align__(16) float          a_s[192];
    __shared__ __align__(16) unsigned short hs1[64 * 64];
    __shared__ __align__(16) unsigned short hs2[64 * 128];
    __shared__ __align__(16) unsigned short hs3[256 * 64];  // [k][m]
    __shared__ __align__(16) unsigned short vt[32 * 64];    // [i][n]

    const int seg  = blockIdx.x;
    const int b    = blockIdx.y;
    const int tid  = threadIdx.x;
    const int wave = tid >> 5;
    const int lane = tid & 31;
    const int half = lane >> 4;
    const int lm   = lane & 15;

    v8f acc[2][2] = {};

    for (int ch = seg; ch < NCHUNK; ch += NSEG) {
        int n0 = ch * 64;
        int P  = NPTS - n0; if (P > 64) P = 64;

        // prefetch next chunk's inputs (overlaps HBM with WMMA work below)
        int chn = ch + NSEG;
        if (chn < NCHUNK) {
            const float* pv = v + ((size_t)b * NPTS + chn * 64) * 32;
            __builtin_prefetch(pv + tid * 32, 0, 1);   // 64x32 f32 tile
            const float* pa = a + ((size_t)b * NPTS + chn * 64) * 3;
            __builtin_prefetch(pa + tid, 0, 1);        // 64x3 f32 tile
        }

        // stage v tile transposed [i][n], zero-padded
        for (int e = tid; e < 2048; e += 256) {
            int n = e >> 5, i = e & 31;
            float val = (n < P) ? v[((size_t)b * NPTS + n0 + n) * 32 + i] : 0.0f;
            vt[i * 64 + n] = f2bf(val);
        }

        mlp3_tile<1>(a + ((size_t)b * NPTS + n0) * 3, P,
                     w1, b1, w2t, b2, w3t, b3, a_s, hs1, hs2, hs3);

        // G tile GEMM: M=k(256: 16 tiles, 2/wave), N=i(32: 2 tiles), K=n(64)
        for (int ks = 0; ks < 2; ++ks) {
            v16bf A0 = load_frag_A(hs3 + ((wave * 2 + 0) * 16 + lm) * 64 + ks * 32, half);
            v16bf A1 = load_frag_A(hs3 + ((wave * 2 + 1) * 16 + lm) * 64 + ks * 32, half);
            v16bf B0 = load_frag_B(vt + (lm) * 64 + ks * 32, half);
            v16bf B1 = load_frag_B(vt + (16 + lm) * 64 + ks * 32, half);
            acc[0][0] = wmma_bf16(A0, B0, acc[0][0]);
            acc[0][1] = wmma_bf16(A0, B1, acc[0][1]);
            acc[1][0] = wmma_bf16(A1, B0, acc[1][0]);
            acc[1][1] = wmma_bf16(A1, B1, acc[1][1]);
        }
        __syncthreads();   // protect vt/hs3 before next chunk overwrites
    }

    float* Gp = Gpart + ((size_t)(seg * BATCH + b)) * 256 * 32;
    for (int mi = 0; mi < 2; ++mi)
        for (int ni = 0; ni < 2; ++ni)
            for (int j = 0; j < 8; ++j) {
                int k = (wave * 2 + mi) * 16 + half * 8 + j;
                int i = ni * 16 + lm;
                Gp[k * 32 + i] = acc[mi][ni][j];
            }
}

// ---------------------------------------------------------------------------
// K3a: deterministic reduce of G partials
// ---------------------------------------------------------------------------
__global__ __launch_bounds__(256) void k3_reduceG(
    const float* __restrict__ Gpart, float* __restrict__ G)
{
    int e = blockIdx.x * 256 + threadIdx.x;
    if (e >= BATCH * 256 * 32) return;
    int b = e / (256 * 32);
    int ki = e - b * (256 * 32);
    float s = 0.0f;
    for (int seg = 0; seg < NSEG; ++seg)
        s += Gpart[((size_t)(seg * BATCH + b)) * 256 * 32 + ki];
    G[e] = s;
}

// ---------------------------------------------------------------------------
// K3b: t[b,o,r] = sum_{k,i} G[b,k,i]*W4psi[k,o*64+i*2+r] + sum_i Vs[b,i]*b4psi
// ---------------------------------------------------------------------------
__global__ __launch_bounds__(256) void k3_t(
    const float* __restrict__ G, const float* __restrict__ Vs,
    const float* __restrict__ psi_w4, const float* __restrict__ psi_b4,
    float* __restrict__ tbuf)
{
    int e = blockIdx.x * 256 + threadIdx.x;
    if (e >= BATCH * 64) return;
    int b = e >> 6, orr = e & 63;
    int o = orr >> 1, r = orr & 1;
    int col0 = o * 64 + r;
    float s = 0.0f;
    for (int k = 0; k < 256; ++k) {
        const float* wrow = psi_w4 + (size_t)k * 2048 + col0;
        const float* grow = G + ((size_t)b * 256 + k) * 32;
        for (int i = 0; i < 32; ++i) s += grow[i] * wrow[i * 2];
    }
    for (int i = 0; i < 32; ++i) s += Vs[b * 32 + i] * psi_b4[col0 + i * 2];
    tbuf[e] = s;
}

// ---------------------------------------------------------------------------
// K3c: Weff_t[b][o][k] (bf16) and beff[b][o]
// ---------------------------------------------------------------------------
__global__ __launch_bounds__(256) void k3_weff(
    const float* __restrict__ tbuf, const float* __restrict__ wsum,
    const float* __restrict__ bsum,
    unsigned short* __restrict__ wefft, float* __restrict__ beff)
{
    int e = blockIdx.x * 256 + threadIdx.x;
    if (e < BATCH * 256 * 32) {
        int b = e >> 13;
        int k = (e >> 5) & 255;
        int o = e & 31;
        float s = tbuf[b * 64 + o * 2 + 0] * wsum[k * 64 + o * 2 + 0]
                + tbuf[b * 64 + o * 2 + 1] * wsum[k * 64 + o * 2 + 1];
        wefft[((size_t)b * 32 + o) * 256 + k] = f2bf(s);
    } else if (e < BATCH * 256 * 32 + BATCH * 32) {
        int e2 = e - BATCH * 256 * 32;
        int b = e2 >> 5, o = e2 & 31;
        beff[e2] = tbuf[b * 64 + o * 2 + 0] * bsum[o * 2 + 0]
                 + tbuf[b * 64 + o * 2 + 1] * bsum[o * 2 + 1];
    }
}

// ---------------------------------------------------------------------------
// K4: fused phi MLP3 + out[b,m,o] = (h3 @ Weff_b + beff_b) / N
// grid (NCHUNK, BATCH), 256 threads
// ---------------------------------------------------------------------------
__global__ __launch_bounds__(256) void k4_phi(
    const float* __restrict__ a,
    const float* __restrict__ w1, const float* __restrict__ b1,
    const unsigned short* __restrict__ w2t, const float* __restrict__ b2,
    const unsigned short* __restrict__ w3t, const float* __restrict__ b3,
    const unsigned short* __restrict__ wefft, const float* __restrict__ beff,
    float* __restrict__ out)
{
    __shared__ __align__(16) float          a_s[192];
    __shared__ __align__(16) unsigned short hs1[64 * 64];
    __shared__ __align__(16) unsigned short hs2[64 * 128];
    __shared__ __align__(16) unsigned short hs3[64 * 256];  // [m][k]

    const int b    = blockIdx.y;
    const int n0   = blockIdx.x * 64;
    int P = NPTS - n0; if (P > 64) P = 64;

    const int tid  = threadIdx.x;
    const int wave = tid >> 5;
    const int lane = tid & 31;
    const int half = lane >> 4;
    const int lm   = lane & 15;

    mlp3_tile<0>(a + ((size_t)b * NPTS + n0) * 3, P,
                 w1, b1, w2t, b2, w3t, b3, a_s, hs1, hs2, hs3);

    // out GEMM: M=m(64: 4 tiles), N=o(32: 2 tiles), 8 waves -> 1 tile each, K=256
    int mi = wave >> 1, ni = wave & 1;
    v8f acc = {};
    const unsigned short* wB = wefft + ((size_t)b * 32 + ni * 16 + lm) * 256;
    for (int ks = 0; ks < 8; ++ks) {
        v16bf A = load_frag_A(hs3 + (mi * 16 + lm) * 256 + ks * 32, half);
        v16bf B = load_frag_B(wB + ks * 32, half);
        acc = wmma_bf16(A, B, acc);
    }
    int o = ni * 16 + lm;
    float bo = beff[b * 32 + o];
    const float scale = 1.0f / (float)NPTS;
    for (int j = 0; j < 8; ++j) {
        int m = mi * 16 + half * 8 + j;
        if (m < P)
            out[((size_t)b * NPTS + n0 + m) * 32 + o] = (acc[j] + bo) * scale;
    }
}

// ---------------------------------------------------------------------------
// Host launch
// ---------------------------------------------------------------------------
extern "C" void kernel_launch(void* const* d_in, const int* in_sizes, int n_in,
                              void* d_out, int out_size, void* d_ws, size_t ws_size,
                              hipStream_t stream) {
    const float* v      = (const float*)d_in[0];
    const float* a      = (const float*)d_in[1];
    const float* phi_w1 = (const float*)d_in[2];
    const float* phi_b1 = (const float*)d_in[3];
    const float* phi_w2 = (const float*)d_in[4];
    const float* phi_b2 = (const float*)d_in[5];
    const float* phi_w3 = (const float*)d_in[6];
    const float* phi_b3 = (const float*)d_in[7];
    const float* phi_w4 = (const float*)d_in[8];
    const float* phi_b4 = (const float*)d_in[9];
    const float* psi_w1 = (const float*)d_in[10];
    const float* psi_b1 = (const float*)d_in[11];
    const float* psi_w2 = (const float*)d_in[12];
    const float* psi_b2 = (const float*)d_in[13];
    const float* psi_w3 = (const float*)d_in[14];
    const float* psi_b3 = (const float*)d_in[15];
    const float* psi_w4 = (const float*)d_in[16];
    const float* psi_b4 = (const float*)d_in[17];
    float* out = (float*)d_out;

    // workspace carve-up (all offsets 256B aligned)
    char* ws = (char*)d_ws;
    unsigned short* w2t_phi = (unsigned short*)(ws + 0);        // 16 KB
    unsigned short* w2t_psi = (unsigned short*)(ws + 16384);    // 16 KB
    unsigned short* w3t_phi = (unsigned short*)(ws + 32768);    // 64 KB
    unsigned short* w3t_psi = (unsigned short*)(ws + 98304);    // 64 KB
    float*          wsum    = (float*)(ws + 163840);            // 64 KB
    float*          bsum    = (float*)(ws + 229376);            // 256 B
    float*          Vs      = (float*)(ws + 229632);            // 1.25 KB
    float*          tbuf    = (float*)(ws + 230912);            // 2.5 KB
    float*          beff    = (float*)(ws + 233472);            // 1.25 KB
    unsigned short* wefft   = (unsigned short*)(ws + 234752);   // 160 KB
    float*          G       = (float*)(ws + 398592);            // 320 KB
    float*          Gpart   = (float*)(ws + 726272);            // NSEG*320 KB
    (void)in_sizes; (void)n_in; (void)out_size; (void)ws_size;

    // K0: prep (98688 work items)
    k0_prep<<<386, 256, 0, stream>>>(phi_w2, psi_w2, phi_w3, psi_w3,
                                     phi_w4, phi_b4, v,
                                     w2t_phi, w2t_psi, w3t_phi, w3t_psi,
                                     wsum, bsum, Vs);

    // K2: fused psi MLP3 + partial G
    k2_psi<<<dim3(NSEG, BATCH), 256, 0, stream>>>(v, a, psi_w1, psi_b1,
                                                  w2t_psi, psi_b2,
                                                  w3t_psi, psi_b3, Gpart);

    // K3: reduce G, compute t, compute Weff/beff
    k3_reduceG<<<(BATCH * 256 * 32 + 255) / 256, 256, 0, stream>>>(Gpart, G);
    k3_t<<<(BATCH * 64 + 255) / 256, 256, 0, stream>>>(G, Vs, psi_w4, psi_b4, tbuf);
    k3_weff<<<(BATCH * 256 * 32 + BATCH * 32 + 255) / 256, 256, 0, stream>>>(
        tbuf, wsum, bsum, wefft, beff);

    // K4: fused phi MLP3 + output GEMM
    k4_phi<<<dim3(NCHUNK, BATCH), 256, 0, stream>>>(a, phi_w1, phi_b1,
                                                    w2t_phi, phi_b2,
                                                    w3t_phi, phi_b3,
                                                    wefft, beff, out);
}
